// ModulatedConv2d_1803886264382
// MI455X (gfx1250) — matrix operand order repaired and checked by reading, not compiled, not run on hardware
//
#include <hip/hip_runtime.h>
#include <hip/hip_bf16.h>
#include <math.h>

// ---------------------------------------------------------------------------
// ModulatedConv2d (StyleGAN2) for gfx1250 / MI455X
//   B=8, C_in=C_out=256, H=W=128, K=3, S=512
//   Compute-bound (~155 GFLOP vs ~270 MB) -> bf16 WMMA path, fp32 accumulate.
//
// v2: stage all 9 taps' A-tiles per ci-chunk (72KB LDS, CDNA5 has 320KB/WGP)
//     -> 2 barriers per chunk instead of 18; wave tile rebalanced to
//     32co x 64px so each B fragment feeds 2 wmmas (1.5 ds_b128 per wmma).
// ---------------------------------------------------------------------------

typedef __attribute__((ext_vector_type(16))) __bf16 v16bf;
typedef __attribute__((ext_vector_type(8)))  float  v8f;

#define B_SZ   8
#define CIN    256
#define COUT   256
#define H_DIM  128
#define W_DIM  128
#define S_DIM  512
#define NCHUNK (CIN / 32)   // 8 ci-chunks of 32 (WMMA bf16 K depth)
#define COLS   132          // 128 px + halo(2) + pad, keeps 16B alignment

__device__ __forceinline__ unsigned short f2bf(float f) {
    // float -> bf16 round-to-nearest-even
    unsigned int u = __float_as_uint(f);
    u += 0x7FFFu + ((u >> 16) & 1u);
    return (unsigned short)(u >> 16);
}

// ---------------------------------------------------------------------------
// Kernel 1: s[b][ci] = style[b] . (mod_weight[ci] * 1/sqrt(S)) + mod_bias[ci]
// ---------------------------------------------------------------------------
__global__ void mod_style_kernel(const float* __restrict__ style,
                                 const float* __restrict__ mod_weight,
                                 const float* __restrict__ mod_bias,
                                 float* __restrict__ s_buf) {
    int idx = blockIdx.x * blockDim.x + threadIdx.x;   // 2048 = B*CIN
    int b  = idx >> 8;
    int ci = idx & 255;
    const float* st = style + (size_t)b * S_DIM;
    const float* mw = mod_weight + (size_t)ci * S_DIM;
    float acc = 0.f;
    for (int k = 0; k < S_DIM; k += 4) {
        float4 a = *(const float4*)(st + k);
        float4 w = *(const float4*)(mw + k);
        acc += a.x * w.x + a.y * w.y + a.z * w.z + a.w * w.w;
    }
    const float lin_scale = 0.04419417382415922f;      // 1/sqrt(512)
    s_buf[idx] = acc * lin_scale + mod_bias[ci];
}

// ---------------------------------------------------------------------------
// Kernel 2: modulate + demodulate weights, store bf16 in GEMM-friendly tiles:
//   Wb[b][cc][tap][co][ci_local]  (each (b,cc,tap) tile over all 256 co is
//   contiguous; a block's 128-co half-tile is a contiguous 8KB blob)
// One block per (co, b): 256 threads, thread = ci.
// ---------------------------------------------------------------------------
__global__ void mod_weight_kernel(const float* __restrict__ weight,
                                  const float* __restrict__ s_buf,
                                  unsigned short* __restrict__ Wb) {
    const int co = blockIdx.x;
    const int b  = blockIdx.y;
    const int ci = threadIdx.x;

    const float conv_scale = 1.0f / 48.0f;             // 1/sqrt(256*9)
    float m = conv_scale * s_buf[b * CIN + ci];

    const float* wp = weight + ((size_t)co * CIN + ci) * 9;
    float w9[9];
    float ss = 0.f;
#pragma unroll
    for (int j = 0; j < 9; ++j) {
        w9[j] = wp[j] * m;
        ss += w9[j] * w9[j];
    }

    __shared__ float red[256];
    red[ci] = ss;
    __syncthreads();
    for (int off = 128; off > 0; off >>= 1) {
        if (ci < off) red[ci] += red[ci + off];
        __syncthreads();
    }
    float demod = rsqrtf(red[0] + 1e-8f);

    const int cc  = ci >> 5;
    const int cil = ci & 31;
#pragma unroll
    for (int t = 0; t < 9; ++t) {
        size_t idx = ((((size_t)b * NCHUNK + cc) * 9 + t) * COUT + co) * 32 + cil;
        Wb[idx] = f2bf(w9[t] * demod);
    }
}

// ---------------------------------------------------------------------------
// Kernel 3: grouped (per-batch) 3x3 conv as implicit GEMM on bf16 WMMA.
// Block = 256 threads (8 wave32). Block tile: 128 out-ch x one output row
// (128 px). Wave tile: 32 co x 64 px (4 co-groups x 2 px-halves).
// Per ci-chunk: stage 3 halo rows (25KB) + all 9 tap A-tiles (72KB) in LDS,
// then 9 taps x 8 wmmas per wave with zero barriers in the compute phase.
// ---------------------------------------------------------------------------
__global__ __launch_bounds__(256) void modconv_wmma_kernel(
        const float* __restrict__ in,
        const unsigned short* __restrict__ Wb,
        float* __restrict__ out) {
    // input stage: [row 0..2][col 0..131][ci 0..31] bf16 (ci contiguous)
    __shared__ __align__(16) unsigned short in_s[3 * COLS * 32];     // 25344 B
    // A stage: [tap 0..8][co_local 0..127][ci 0..31] bf16             73728 B
    __shared__ __align__(16) unsigned short a_all[9 * 128 * 32];

    const int tid  = threadIdx.x;
    const int wave = tid >> 5;
    const int lane = tid & 31;
    const int l    = lane & 15;
    const int hi   = lane >> 4;            // lane group 0/1

    const int cg = wave >> 1;              // co-group 0..3  -> co_local cg*32
    const int ph = wave & 1;               // px-half 0..1   -> px base ph*64

    const int y   = blockIdx.x;            // output row
    const int co0 = blockIdx.y * 128;      // output channel tile base
    const int b   = blockIdx.z;

    v8f acc[2][4] = {};                    // [co-frag 0..1][px-group 0..3]

    const float* in_b = in + (size_t)b * CIN * H_DIM * W_DIM;

    for (int kc = 0; kc < NCHUNK; ++kc) {
        __syncthreads();                   // protect LDS reuse across chunks

        // ---- stage input: 3 halo rows x 132 cols x 32 ci, fp32 -> bf16 ----
        for (int i = tid; i < 32 * 3 * COLS; i += 256) {
            int ci  = i / (3 * COLS);
            int rem = i - ci * (3 * COLS);
            int r   = rem / COLS;
            int col = rem - r * COLS;
            int iy  = y + r - 1;
            int ix  = col - 1;
            float v = 0.f;
            if ((unsigned)iy < H_DIM && (unsigned)ix < W_DIM)
                v = in_b[((size_t)(kc * 32 + ci) * H_DIM + iy) * W_DIM + ix];
            in_s[(r * COLS + col) * 32 + ci] = f2bf(v);
        }

        // ---- stage all 9 tap A-tiles: 9 x (128 co x 32 ci) bf16 = 72KB ----
        {
            const unsigned short* wsrc =
                Wb + (size_t)(b * NCHUNK + kc) * 9 * COUT * 32;
            uint4* dst = (uint4*)a_all;
            for (int i = tid; i < 9 * 512; i += 256) {   // 512 uint4 per tap
                int t = i >> 9;
                int r = i & 511;
                const uint4* src =
                    (const uint4*)(wsrc + ((size_t)t * COUT + co0) * 32);
                dst[i] = src[r];
            }
        }

        // ---- prefetch next chunk's input rows (global_prefetch_b8) ----
        if (kc + 1 < NCHUNK) {
            for (int i = tid; i < 32 * 3; i += 256) {
                int ci = i / 3;
                int r  = i - ci * 3;
                int iy = y + r - 1;
                if ((unsigned)iy < H_DIM)
                    __builtin_prefetch(
                        &in_b[((size_t)((kc + 1) * 32 + ci) * H_DIM + iy) * W_DIM],
                        0, 1);
            }
        }

        __syncthreads();

        // ---- compute: 9 taps, no barriers ----
#pragma unroll
        for (int t = 0; t < 9; ++t) {
            const int ky = t / 3;
            const int kx = t - ky * 3;

            // A fragments (ISA 16-bit A 16x32 layout):
            // lanes 0-15: M=lane, K 0-7 & 16-23; lanes 16-31: M=lane-16, K 8-15 & 24-31
            union { uint4 u[2]; v16bf v; } af0, af1;
            {
                const uint4* ap0 = (const uint4*)(a_all +
                    (((t * 128) + cg * 32 + l) * 32 + hi * 8));
                af0.u[0] = ap0[0];
                af0.u[1] = ap0[2];
                const uint4* ap1 = (const uint4*)(a_all +
                    (((t * 128) + cg * 32 + 16 + l) * 32 + hi * 8));
                af1.u[0] = ap1[0];
                af1.u[1] = ap1[2];
            }

#pragma unroll
            for (int pg = 0; pg < 4; ++pg) {
                // B fragment: lanes 0-15 -> N=lane, K 0-7 & 16-23;
                //             lanes 16-31 -> N=lane-16, K 8-15 & 24-31
                union { uint4 u[2]; v16bf v; } bfrag;
                const int col = ph * 64 + pg * 16 + l + kx;  // tap-shifted
                const uint4* bp =
                    (const uint4*)(in_s + ((ky * COLS + col) * 32 + hi * 8));
                bfrag.u[0] = bp[0];
                bfrag.u[1] = bp[2];

                acc[0][pg] = __builtin_amdgcn_wmma_f32_16x16x32_bf16(
                    false, af0.v, false, bfrag.v,
                    (short)0, acc[0][pg], false, false);
                acc[1][pg] = __builtin_amdgcn_wmma_f32_16x16x32_bf16(
                    false, af1.v, false, bfrag.v,
                    (short)0, acc[1][pg], false, false);
            }
        }
    }

    // ---- epilogue: C/D layout -> out[b][co][y][x] fp32 ----
    float* out_b = out + ((size_t)b * COUT + co0) * H_DIM * W_DIM;
#pragma unroll
    for (int af = 0; af < 2; ++af) {
#pragma unroll
        for (int pg = 0; pg < 4; ++pg) {
#pragma unroll
            for (int v = 0; v < 8; ++v) {
                int co_l = cg * 32 + af * 16 + hi * 8 + v;  // M = v (+8 hi lanes)
                int x    = ph * 64 + pg * 16 + l;           // N = lane % 16
                out_b[((size_t)co_l * H_DIM + y) * W_DIM + x] = acc[af][pg][v];
            }
        }
    }
}

// ---------------------------------------------------------------------------
// Launch
// ---------------------------------------------------------------------------
extern "C" void kernel_launch(void* const* d_in, const int* in_sizes, int n_in,
                              void* d_out, int out_size, void* d_ws, size_t ws_size,
                              hipStream_t stream) {
    const float* input      = (const float*)d_in[0]; // [8,256,128,128]
    const float* style      = (const float*)d_in[1]; // [8,512]
    const float* weight     = (const float*)d_in[2]; // [1,256,256,3,3]
    const float* mod_weight = (const float*)d_in[3]; // [256,512]
    const float* mod_bias   = (const float*)d_in[4]; // [256]
    float* out = (float*)d_out;                      // [8,256,128,128]

    // workspace: s_buf (8KB fp32) | Wb (9.4MB bf16 tiled weights)
    float* s_buf = (float*)d_ws;
    unsigned short* Wb = (unsigned short*)((char*)d_ws + 8192);

    mod_style_kernel<<<8, 256, 0, stream>>>(style, mod_weight, mod_bias, s_buf);
    mod_weight_kernel<<<dim3(COUT, B_SZ), 256, 0, stream>>>(weight, s_buf, Wb);
    modconv_wmma_kernel<<<dim3(H_DIM, COUT / 128, B_SZ), 256, 0, stream>>>(
        input, Wb, out);
}
